// MultiHeadAttention_28819230556860
// MI455X (gfx1250) — matrix operand order repaired and checked by reading, not compiled, not run on hardware
//
#include <hip/hip_runtime.h>
#include <hip/hip_bf16.h>

#define Bn 4
#define Tn 2048
#define Cn 768
#define Hn 12
#define Dn 64

typedef __bf16 bf16_t;
typedef __attribute__((ext_vector_type(16))) __bf16 v16bf;
typedef __attribute__((ext_vector_type(8))) float v8f;

// ---------------------------------------------------------------------------
// WMMA helpers (CDNA5 gfx1250, wave32): D = A(16x32 bf16) * B(32x16 bf16) + C(f32)
// ---------------------------------------------------------------------------
__device__ __forceinline__ v8f wmma_bf16(v16bf a, v16bf b, v8f c) {
  return __builtin_amdgcn_wmma_f32_16x16x32_bf16(
      /*neg_a=*/false, a, /*neg_b=*/false, b,
      /*c_mod=*/(short)0, c, /*reuse_a=*/false, /*reuse_b=*/false);
}

__device__ __forceinline__ v8f zero8() {
  v8f z;
  for (int i = 0; i < 8; ++i) z[i] = 0.0f;
  return z;
}

// A-matrix 16x32 bf16 fragment from row-major storage [row][k], stride in elems.
// Lanes 0-15: M=lane, K = {0..7, 16..23}; lanes 16-31: M=lane-16, (K base 8/24).
__device__ __forceinline__ v16bf load_a_frag(const bf16_t* base, int stride, int lane) {
  const int row = lane & 15;
  const int kb = (lane < 16) ? 0 : 8;
  const bf16_t* p = base + row * stride + kb;
  v16bf a;
#pragma unroll
  for (int e = 0; e < 8; ++e) {
    a[e] = p[e];
    a[8 + e] = p[16 + e];
  }
  return a;
}

// B-matrix 32x16 bf16 fragment from N-major (transposed) storage [n][k].
// Lanes 0-15: N=lane, K={0..7,16..23}; lanes 16-31: N=lane-16, K={8..15,24..31}.
__device__ __forceinline__ v16bf load_b_frag(const bf16_t* base, int stride, int lane) {
  const int n = lane & 15;
  const int kb = (lane < 16) ? 0 : 8;
  const bf16_t* p = base + n * stride + kb;
  v16bf b;
#pragma unroll
  for (int e = 0; e < 16; ++e) b[e] = p[(e < 8) ? e : (8 + e)];
  return b;
}

// ---------------------------------------------------------------------------
// K1: q/k/v = x @ W[h]   (M=128 tokens, N=64, K=768)  -> bf16 [B,H,T,D]
// grid: (B*T/128, 3*H), block: 256 (8 waves, each 32x32 output tile)
// ---------------------------------------------------------------------------
__global__ __launch_bounds__(256) void qkv_kernel(
    const float* __restrict__ x, const float* __restrict__ Wq,
    const float* __restrict__ Wk, const float* __restrict__ Wv,
    bf16_t* __restrict__ qb, bf16_t* __restrict__ kb, bf16_t* __restrict__ vb) {
  const int mblk = blockIdx.x;
  const int proj = blockIdx.y / Hn;  // 0=q,1=k,2=v
  const int h = blockIdx.y % Hn;
  const int bt0 = mblk * 128;
  const int b = bt0 / Tn;
  const int t0 = bt0 % Tn;
  const float* W = (proj == 0 ? Wq : (proj == 1 ? Wk : Wv)) + (size_t)h * Cn * Dn;
  bf16_t* out = (proj == 0 ? qb : (proj == 1 ? kb : vb)) +
                (((size_t)b * Hn + h) * Tn + t0) * Dn;

  __shared__ bf16_t xs[128][40];  // x tile, row-major [m][k]
  __shared__ bf16_t wt[64][40];   // W tile, N-major   [n][k]

  const int tid = threadIdx.x;
  const int lane = tid & 31;
  const int wid = tid >> 5;
  const int wm = wid >> 1;  // 0..3 -> M offset wm*32
  const int wn = wid & 1;   // 0..1 -> N offset wn*32

  v8f acc[2][2];
#pragma unroll
  for (int i = 0; i < 2; ++i)
#pragma unroll
    for (int j = 0; j < 2; ++j) acc[i][j] = zero8();

  for (int k0 = 0; k0 < Cn; k0 += 32) {
    __syncthreads();
    for (int i = tid; i < 128 * 32; i += 256) {
      const int r = i >> 5, c = i & 31;
      xs[r][c] = (bf16_t)x[((size_t)bt0 + r) * Cn + k0 + c];
    }
    for (int i = tid; i < 64 * 32; i += 256) {
      const int n = i & 63, kk = i >> 6;
      wt[n][kk] = (bf16_t)W[(size_t)(k0 + kk) * Dn + n];
    }
    __syncthreads();
    const v16bf a0 = load_a_frag(&xs[wm * 32][0], 40, lane);
    const v16bf a1 = load_a_frag(&xs[wm * 32 + 16][0], 40, lane);
    const v16bf b0 = load_b_frag(&wt[wn * 32][0], 40, lane);
    const v16bf b1 = load_b_frag(&wt[wn * 32 + 16][0], 40, lane);
    acc[0][0] = wmma_bf16(a0, b0, acc[0][0]);
    acc[1][0] = wmma_bf16(a1, b0, acc[1][0]);
    acc[0][1] = wmma_bf16(a0, b1, acc[0][1]);
    acc[1][1] = wmma_bf16(a1, b1, acc[1][1]);
  }

  const int rhi = (lane < 16) ? 0 : 8;
#pragma unroll
  for (int mt = 0; mt < 2; ++mt)
#pragma unroll
    for (int nt = 0; nt < 2; ++nt) {
      const int n = wn * 32 + nt * 16 + (lane & 15);
      const int m0 = wm * 32 + mt * 16 + rhi;
#pragma unroll
      for (int r = 0; r < 8; ++r)
        out[(size_t)(m0 + r) * Dn + n] = (bf16_t)acc[mt][nt][r];
    }
}

// ---------------------------------------------------------------------------
// K2: per-(b,h,d) L2 norm over the TOKEN axis (torch F.normalize dim=1).
// qinv = (C^-0.5)/max(||q_col||,1e-12), kinv = 1/max(||k_col||,1e-12)
// grid: (B*H), block 256 (4 row-groups x 64 columns)
// ---------------------------------------------------------------------------
__global__ __launch_bounds__(256) void norm_kernel(
    const bf16_t* __restrict__ qb, const bf16_t* __restrict__ kb,
    float* __restrict__ qinv, float* __restrict__ kinv) {
  const int bh = blockIdx.x;
  const bf16_t* qh = qb + (size_t)bh * Tn * Dn;
  const bf16_t* kh = kb + (size_t)bh * Tn * Dn;
  const int tid = threadIdx.x;
  const int d = tid & 63;
  const int g = tid >> 6;  // 0..3
  float sq = 0.0f, sk = 0.0f;
  for (int t = g; t < Tn; t += 4) {
    const float a = (float)qh[(size_t)t * Dn + d];
    const float c = (float)kh[(size_t)t * Dn + d];
    sq += a * a;
    sk += c * c;
  }
  __shared__ float redq[4][64];
  __shared__ float redk[4][64];
  redq[g][d] = sq;
  redk[g][d] = sk;
  __syncthreads();
  if (tid < 64) {
    const float tq = redq[0][tid] + redq[1][tid] + redq[2][tid] + redq[3][tid];
    const float tk = redk[0][tid] + redk[1][tid] + redk[2][tid] + redk[3][tid];
    const float scale = 0.03608439182435161f;  // 768^-0.5 folded into q
    qinv[(size_t)bh * Dn + tid] = scale / fmaxf(sqrtf(tq), 1e-12f);
    kinv[(size_t)bh * Dn + tid] = 1.0f / fmaxf(sqrtf(tk), 1e-12f);
  }
}

// ---------------------------------------------------------------------------
// K3: causal softmax-free attention, flash-style 128x128 tiles.
// Each wave owns 16 query rows and the full 128 key columns of the score tile,
// so O = tril(W) @ V needs no cross-wave reduction. Score accumulators are
// transposed C/D->A layout through a 16x32 per-wave LDS scratch.
// grid: (T/128, B*H), block 256 (8 waves)
// ---------------------------------------------------------------------------
__global__ __launch_bounds__(256) void attn_kernel(
    const bf16_t* __restrict__ qb, const bf16_t* __restrict__ kb,
    const bf16_t* __restrict__ vb, const float* __restrict__ qinv,
    const float* __restrict__ kinv, bf16_t* __restrict__ att) {
  const int qc = blockIdx.x;  // query chunk
  const int bh = blockIdx.y;
  const int b = bh / Hn, h = bh % Hn;
  const size_t base = (size_t)bh * Tn * Dn;
  const bf16_t* qh = qb + base;
  const bf16_t* kh = kb + base;
  const bf16_t* vh = vb + base;
  const float* qiv = qinv + (size_t)bh * Dn;
  const float* kiv = kinv + (size_t)bh * Dn;
  const int t0 = qc * 128;

  __shared__ bf16_t ks[128][66];       // normalized k chunk, [s][d]
  __shared__ bf16_t vst[64][132];      // v chunk transposed, [d][s]
  __shared__ bf16_t wscr[8][16][36];   // per-wave 16x32 score transpose scratch

  const int tid = threadIdx.x, lane = tid & 31, wid = tid >> 5;  // wave: 16 rows

  // q A-fragments, loaded once with normalization (and C^-0.5) folded in.
  v16bf qa0, qa1;
  {
    const int arow = wid * 16 + (lane & 15);
    const int akb = (lane < 16) ? 0 : 8;
    const bf16_t* qrow = qh + (size_t)(t0 + arow) * Dn;
#pragma unroll
    for (int e = 0; e < 8; ++e) {
      qa0[e]     = (bf16_t)((float)qrow[akb + e] * qiv[akb + e]);
      qa0[8 + e] = (bf16_t)((float)qrow[akb + 16 + e] * qiv[akb + 16 + e]);
      qa1[e]     = (bf16_t)((float)qrow[32 + akb + e] * qiv[32 + akb + e]);
      qa1[8 + e] = (bf16_t)((float)qrow[32 + akb + 16 + e] * qiv[32 + akb + 16 + e]);
    }
  }

  v8f oacc[4];
#pragma unroll
  for (int nt = 0; nt < 4; ++nt) oacc[nt] = zero8();

  for (int sc = 0; sc <= qc; ++sc) {
    const int s0 = sc * 128;
    __syncthreads();
    for (int i = tid; i < 128 * 64; i += 256) {
      const int r = i >> 6, d = i & 63;
      ks[r][d] = (bf16_t)((float)kh[(size_t)(s0 + r) * Dn + d] * kiv[d]);
      vst[d][r] = vh[(size_t)(s0 + r) * Dn + d];
    }
    if (sc < qc)  // pull next k chunk toward the caches (global_prefetch_b8)
      __builtin_prefetch(&kh[(size_t)(s0 + 128) * Dn + ((size_t)tid << 5)], 0, 1);
    __syncthreads();

    // W = qn @ kn^T : this wave's 16 rows x all 128 columns.
    v8f wacc[8];
#pragma unroll
    for (int nt = 0; nt < 8; ++nt) wacc[nt] = zero8();
#pragma unroll
    for (int nt = 0; nt < 8; ++nt) {
      const v16bf bk0 = load_b_frag(&ks[nt * 16][0], 66, lane);
      const v16bf bk1 = load_b_frag(&ks[nt * 16][32], 66, lane);
      wacc[nt] = wmma_bf16(qa0, bk0, wacc[nt]);
      wacc[nt] = wmma_bf16(qa1, bk1, wacc[nt]);
    }

    // Mask (diagonal chunk only), transpose 32 K-columns at a time, O += W@V.
    const bool diag = (sc == qc);
    const int rbase = (lane < 16) ? 0 : 8;
    const int mrow = wid * 16 + rbase;  // global row within the 128-block
#pragma unroll
    for (int kc = 0; kc < 4; ++kc) {
#pragma unroll
      for (int j = 0; j < 2; ++j) {
        const int nt = kc * 2 + j;
        const int ncol = nt * 16 + (lane & 15);
#pragma unroll
        for (int r = 0; r < 8; ++r) {
          float w = wacc[nt][r];
          if (diag && ncol > mrow + r) w = 0.0f;  // keep s <= t
          wscr[wid][rbase + r][j * 16 + (lane & 15)] = (bf16_t)w;
        }
      }
      asm volatile("s_wait_dscnt 0" ::: "memory");  // same-wave LDS RAW fence
      const v16bf aw = load_a_frag(&wscr[wid][0][0], 36, lane);
#pragma unroll
      for (int nt2 = 0; nt2 < 4; ++nt2) {
        const v16bf bv = load_b_frag(&vst[nt2 * 16][kc * 32], 132, lane);
        oacc[nt2] = wmma_bf16(aw, bv, oacc[nt2]);
      }
      asm volatile("s_wait_dscnt 0" ::: "memory");  // WAR before scratch reuse
    }
  }

  // Store concat-head output: att[b, t, h*64+d] (bf16)
  const int orow0 = t0 + wid * 16 + ((lane < 16) ? 0 : 8);
#pragma unroll
  for (int nt = 0; nt < 4; ++nt) {
    const int dcol = h * Dn + nt * 16 + (lane & 15);
#pragma unroll
    for (int r = 0; r < 8; ++r)
      att[((size_t)b * Tn + orow0 + r) * Cn + dcol] = (bf16_t)oacc[nt][r];
  }
}

// ---------------------------------------------------------------------------
// K4: y = att @ Wp + bp   (M=8192, N=768, K=768), f32 output.
// grid: (B*T/128, C/64), block 256
// ---------------------------------------------------------------------------
__global__ __launch_bounds__(256) void proj_kernel(
    const bf16_t* __restrict__ att, const float* __restrict__ Wp,
    const float* __restrict__ bp, float* __restrict__ y) {
  const int bt0 = blockIdx.x * 128;
  const int n0 = blockIdx.y * 64;
  __shared__ bf16_t as_[128][40];
  __shared__ bf16_t wt[64][40];
  const int tid = threadIdx.x, lane = tid & 31, wid = tid >> 5;
  const int wm = wid >> 1, wn = wid & 1;

  v8f acc[2][2];
#pragma unroll
  for (int i = 0; i < 2; ++i)
#pragma unroll
    for (int j = 0; j < 2; ++j) acc[i][j] = zero8();

  for (int k0 = 0; k0 < Cn; k0 += 32) {
    __syncthreads();
    for (int i = tid; i < 128 * 32; i += 256) {
      const int r = i >> 5, c = i & 31;
      as_[r][c] = att[((size_t)bt0 + r) * Cn + k0 + c];
    }
    for (int i = tid; i < 64 * 32; i += 256) {
      const int n = i & 63, kk = i >> 6;
      wt[n][kk] = (bf16_t)Wp[(size_t)(k0 + kk) * Cn + n0 + n];
    }
    __syncthreads();
    const v16bf a0 = load_a_frag(&as_[wm * 32][0], 40, lane);
    const v16bf a1 = load_a_frag(&as_[wm * 32 + 16][0], 40, lane);
    const v16bf b0 = load_b_frag(&wt[wn * 32][0], 40, lane);
    const v16bf b1 = load_b_frag(&wt[wn * 32 + 16][0], 40, lane);
    acc[0][0] = wmma_bf16(a0, b0, acc[0][0]);
    acc[1][0] = wmma_bf16(a1, b0, acc[1][0]);
    acc[0][1] = wmma_bf16(a0, b1, acc[0][1]);
    acc[1][1] = wmma_bf16(a1, b1, acc[1][1]);
  }

  const int rhi = (lane < 16) ? 0 : 8;
#pragma unroll
  for (int mt = 0; mt < 2; ++mt)
#pragma unroll
    for (int nt = 0; nt < 2; ++nt) {
      const int n = wn * 32 + nt * 16 + (lane & 15);
      const int m0 = wm * 32 + mt * 16 + rhi;
      const float bias = bp[n0 + n];
#pragma unroll
      for (int r = 0; r < 8; ++r)
        y[((size_t)bt0 + m0 + r) * Cn + n0 + n] = acc[mt][nt][r] + bias;
    }
}

// ---------------------------------------------------------------------------
extern "C" void kernel_launch(void* const* d_in, const int* in_sizes, int n_in,
                              void* d_out, int out_size, void* d_ws, size_t ws_size,
                              hipStream_t stream) {
  const float* x = (const float*)d_in[0];
  const float* Wq = (const float*)d_in[1];
  const float* Wk = (const float*)d_in[2];
  const float* Wv = (const float*)d_in[3];
  const float* Wp = (const float*)d_in[4];
  const float* bp = (const float*)d_in[5];
  float* y = (float*)d_out;

  const size_t nqkv = (size_t)Bn * Hn * Tn * Dn;  // 6,291,456 (== B*T*C)
  char* ws = (char*)d_ws;
  bf16_t* qb = (bf16_t*)(ws);
  bf16_t* kb = (bf16_t*)(ws + nqkv * 2);
  bf16_t* vb = (bf16_t*)(ws + nqkv * 4);
  bf16_t* attb = (bf16_t*)(ws + nqkv * 6);
  float* qinv = (float*)(ws + nqkv * 8);
  float* kinv = qinv + (size_t)Bn * Hn * Dn;

  qkv_kernel<<<dim3((Bn * Tn) / 128, 3 * Hn), 256, 0, stream>>>(x, Wq, Wk, Wv, qb, kb, vb);
  norm_kernel<<<dim3(Bn * Hn), 256, 0, stream>>>(qb, kb, qinv, kinv);
  attn_kernel<<<dim3(Tn / 128, Bn * Hn), 256, 0, stream>>>(qb, kb, vb, qinv, kinv, attb);
  proj_kernel<<<dim3((Bn * Tn) / 128, Cn / 64), 256, 0, stream>>>(attb, Wp, bp, y);
}
